// STGCNBlock_52785148068460
// MI455X (gfx1250) — matrix operand order repaired and checked by reading, not compiled, not run on hardware
//
#include <hip/hip_runtime.h>
#include <hip/hip_bf16.h>

// ---------------------------------------------------------------------------
// STGCN block for MI455X (gfx1250, wave32, WMMA).
// Fixed problem size from the reference: B=1, T=64, N=2000, C=64, K=3.
// All GEMM-like contractions run on v_wmma_f32_16x16x32_bf16.
// ---------------------------------------------------------------------------

typedef __attribute__((ext_vector_type(16))) __bf16 v16bf;
typedef __attribute__((ext_vector_type(8)))  float  v8f;

#define TT 64
#define NN 2000
#define CC 64
#define NT_TILES   (NN / 16)            // 125 row-tiles per t
#define ROW_TILES  (TT * NT_TILES)      // 8000 16-row tiles
#define TOTAL_ELEMS (TT * NN * CC)      // 8,192,000
#define BN_M ((float)(TT * NN))         // 128,000 samples per channel

union FragU { uint4 u4[2]; v16bf v; };

// Build a bf16 A-fragment from an f32 row (converting on the fly).
// p points at row_base + kstep*32 + half*8.  ISA 16-bit A layout:
//   element j -> K = (j/8)*16 + half*8 + (j%8)  => runs p[0..7], p[16..23].
__device__ inline v16bf a_frag_from_f32(const float* __restrict__ p) {
  float f[16];
  *(float4*)(f + 0)  = ((const float4*)p)[0];
  *(float4*)(f + 4)  = ((const float4*)p)[1];
  *(float4*)(f + 8)  = ((const float4*)(p + 16))[0];
  *(float4*)(f + 12) = ((const float4*)(p + 16))[1];
  v16bf a;
#pragma unroll
  for (int j = 0; j < 16; ++j) a[j] = (__bf16)f[j];
  return a;
}

__device__ inline v16bf a_frag_from_bf16(const __bf16* __restrict__ p) {
  FragU u;
  u.u4[0] = *(const uint4*)p;          // elements 0..7   (16 B)
  u.u4[1] = *(const uint4*)(p + 16);   // elements 8..15  (16 B)
  return u.v;
}

__device__ inline v16bf b_frag(const __bf16* __restrict__ p) {
  FragU u;                             // 16 contiguous bf16 per lane (32 B)
  u.u4[0] = *(const uint4*)p;
  u.u4[1] = *(const uint4*)(p + 8);
  return u.v;
}

// ---------------------------------------------------------------------------
// Weight packing: lay B matrices out in per-lane fragment order.
// GEMM reads frag (term k, kstep s, coltile nt) at offset
//   ((k*2+s)*4+nt)*512 + lane*16 + j   (bf16 elements).
// Fragment element j at lane L corresponds to
//   K-row  i = s*32 + (j/8)*16 + (L/16)*8 + (j%8) ,  out-col o = nt*16 + (L%16).
// src_io==1 : source indexed [o][i] (conv weights, (O,I,1,K))
// src_io==0 : source indexed [i][o] (gcn_w, (C_in, C_out))
// ---------------------------------------------------------------------------
__global__ __launch_bounds__(256) void pack_kernel(
    const float* __restrict__ w, __bf16* __restrict__ dst, int KT, int src_io) {
  int idx = blockIdx.x * 256 + threadIdx.x;
  if (idx >= KT * 4096) return;
  int j    = idx & 15;
  int lane = (idx >> 4) & 31;
  int frag = idx >> 9;
  int nt = frag & 3, s = (frag >> 2) & 1, k = frag >> 3;
  int i = s * 32 + (j & 8) * 2 + (lane >> 4) * 8 + (j & 7);
  int o = nt * 16 + (lane & 15);
  float v = src_io ? w[(o * CC + i) * KT + k] : w[(i * CC + o) * KT + k];
  dst[idx] = (__bf16)v;
}

// ---------------------------------------------------------------------------
// Temporal conv as 3 accumulated shifted GEMMs.
//   out[(t,n), o] = sum_k sum_i A[(t+k-1, n), i] * Wk[i,o] + bias[o]
// Block = 4 waves; wave w owns row-tile (blockIdx.x*4 + w) = 16 rows x 64 cols.
// ---------------------------------------------------------------------------
template <bool OUT_BF16>
__global__ __launch_bounds__(128) void tconv_kernel(
    const float* __restrict__ A, const __bf16* __restrict__ Wpack,
    const float* __restrict__ bias, void* __restrict__ outv) {
  __shared__ __align__(16) float stage[4][16][64];
  const int wave = threadIdx.x >> 5, lane = threadIdx.x & 31;
  const int tile = blockIdx.x * 4 + wave;          // 0..7999, exact
  const int t  = tile / NT_TILES;                  // wave-uniform
  const int n0 = (tile % NT_TILES) * 16;
  const int half = lane >> 4, rlo = lane & 15;

  v8f acc[4];
#pragma unroll
  for (int nt = 0; nt < 4; ++nt)
#pragma unroll
    for (int r = 0; r < 8; ++r) acc[nt][r] = 0.0f;

  float bvals[4];
#pragma unroll
  for (int nt = 0; nt < 4; ++nt) bvals[nt] = bias[nt * 16 + rlo];

#pragma unroll
  for (int k = 0; k < 3; ++k) {
    const int trow = t + k - 1;                    // uniform branch: EXEC stays full
    if (trow < 0 || trow >= TT) continue;
    const float* arow = A + ((size_t)trow * NN + n0 + rlo) * CC;
#pragma unroll
    for (int s = 0; s < 2; ++s) {
      v16bf a = a_frag_from_f32(arow + s * 32 + half * 8);
#pragma unroll
      for (int nt = 0; nt < 4; ++nt) {
        v16bf b = b_frag(Wpack + (((k * 2 + s) * 4 + nt) * 512 + lane * 16));
        acc[nt] = __builtin_amdgcn_wmma_f32_16x16x32_bf16(
            false, a, false, b, (short)0, acc[nt], false, false);
      }
    }
  }

  // C/D layout: element r -> row r + 8*half, col nt*16 + (lane%16)
#pragma unroll
  for (int nt = 0; nt < 4; ++nt)
#pragma unroll
    for (int r = 0; r < 8; ++r)
      stage[wave][r + 8 * half][nt * 16 + rlo] = acc[nt][r] + bvals[nt];
  __syncthreads();

  const int row = lane >> 1, cb = (lane & 1) * 32;
  const size_t obase = ((size_t)tile * 16 + row) * CC + cb;
  if constexpr (OUT_BF16) {
    __bf16* out = (__bf16*)outv;
#pragma unroll
    for (int chunk = 0; chunk < 2; ++chunk) {
      union { __bf16 h[16]; uint4 u4[2]; } pk;
#pragma unroll
      for (int q = 0; q < 16; ++q)
        pk.h[q] = (__bf16)stage[wave][row][cb + chunk * 16 + q];
      *(uint4*)(out + obase + chunk * 16)     = pk.u4[0];
      *(uint4*)(out + obase + chunk * 16 + 8) = pk.u4[1];
    }
  } else {
    float* out = (float*)outv;
#pragma unroll
    for (int chunk = 0; chunk < 8; ++chunk)
      *(float4*)(out + obase + chunk * 4) =
          *(const float4*)&stage[wave][row][cb + chunk * 4];
  }
}

// ---------------------------------------------------------------------------
// GCN linear: xh[(t,n), d] = sum_c h1[(t,n), c] * gcn_w[c,d]   (bf16 A input)
// ---------------------------------------------------------------------------
__global__ __launch_bounds__(128) void gcn_gemm_kernel(
    const __bf16* __restrict__ h1, const __bf16* __restrict__ Gpack,
    float* __restrict__ xh) {
  __shared__ __align__(16) float stage[4][16][64];
  const int wave = threadIdx.x >> 5, lane = threadIdx.x & 31;
  const int tile = blockIdx.x * 4 + wave;
  const int half = lane >> 4, rlo = lane & 15;

  v8f acc[4];
#pragma unroll
  for (int nt = 0; nt < 4; ++nt)
#pragma unroll
    for (int r = 0; r < 8; ++r) acc[nt][r] = 0.0f;

  const __bf16* arow = h1 + ((size_t)tile * 16 + rlo) * CC;
#pragma unroll
  for (int s = 0; s < 2; ++s) {
    v16bf a = a_frag_from_bf16(arow + s * 32 + half * 8);
#pragma unroll
    for (int nt = 0; nt < 4; ++nt) {
      v16bf b = b_frag(Gpack + ((s * 4 + nt) * 512 + lane * 16));
      acc[nt] = __builtin_amdgcn_wmma_f32_16x16x32_bf16(
          false, a, false, b, (short)0, acc[nt], false, false);
    }
  }

#pragma unroll
  for (int nt = 0; nt < 4; ++nt)
#pragma unroll
    for (int r = 0; r < 8; ++r)
      stage[wave][r + 8 * half][nt * 16 + rlo] = acc[nt][r];
  __syncthreads();

  const int row = lane >> 1, cb = (lane & 1) * 32;
  const size_t obase = ((size_t)tile * 16 + row) * CC + cb;
#pragma unroll
  for (int chunk = 0; chunk < 8; ++chunk)
    *(float4*)(xh + obase + chunk * 4) =
        *(const float4*)&stage[wave][row][cb + chunk * 4];
}

// ---------------------------------------------------------------------------
// Degree / norm helpers (self-loops included: deg starts at 1)
// ---------------------------------------------------------------------------
__global__ __launch_bounds__(256) void deg_init_kernel(float* deg) {
  int n = blockIdx.x * 256 + threadIdx.x;
  if (n < NN) deg[n] = 1.0f;
}
__global__ __launch_bounds__(256) void deg_edge_kernel(
    float* deg, const int* __restrict__ ei, int E) {
  int e = blockIdx.x * 256 + threadIdx.x;
  if (e < E) atomicAdd(&deg[ei[E + e]], 1.0f);
}
__global__ __launch_bounds__(256) void deg_fin_kernel(float* deg) {
  int n = blockIdx.x * 256 + threadIdx.x;
  if (n < NN) deg[n] = rsqrtf(deg[n]);        // deg >= 1 always
}

// agg initialized with gcn bias so scatter lands on top of it.
__global__ __launch_bounds__(256) void agg_init_kernel(
    float* __restrict__ agg, const float* __restrict__ gb) {
  size_t i = (size_t)blockIdx.x * 256 + threadIdx.x;
  if (i < (size_t)TOTAL_ELEMS) agg[i] = gb[i & 63];
}

// Scatter: thread = (edge, channel), loops over T so indices load once.
__global__ __launch_bounds__(256) void scatter_kernel(
    float* __restrict__ agg, const float* __restrict__ xh,
    const int* __restrict__ ei, const float* __restrict__ dinv, int E) {
  int idx = blockIdx.x * 256 + threadIdx.x;
  int tot = (E + NN) * 64;
  if (idx >= tot) return;
  int e = idx >> 6, c = idx & 63;
  int src, dst;
  if (e < E) { src = ei[e]; dst = ei[E + e]; }
  else       { src = dst = e - E; }            // self loop
  float nrm = dinv[src] * dinv[dst];
  const float* xp = xh + (size_t)src * CC + c;
  float* ap = agg + (size_t)dst * CC + c;
  const size_t tstride = (size_t)NN * CC;
#pragma unroll 4
  for (int t = 0; t < TT; ++t)
    atomicAdd(ap + (size_t)t * tstride, xp[(size_t)t * tstride] * nrm);
}

// ---------------------------------------------------------------------------
// BatchNorm stats (per channel over B*N*T), then normalize+ReLU in place.
// ---------------------------------------------------------------------------
__global__ __launch_bounds__(128) void stats_init_kernel(float* stats) {
  if (threadIdx.x < 128) stats[threadIdx.x] = 0.0f;
}
__global__ __launch_bounds__(256) void stats_kernel(
    const float* __restrict__ h, float* __restrict__ stats) {
  __shared__ float s1s[256], s2s[256];
  int tid = threadIdx.x;
  // step is a multiple of 64 => every thread sees a fixed channel (tid & 63)
  size_t i = (size_t)blockIdx.x * 256 + tid;
  const size_t step = (size_t)512 * 256;
  float s1 = 0.0f, s2 = 0.0f;
  for (; i < (size_t)TOTAL_ELEMS; i += step) { float v = h[i]; s1 += v; s2 += v * v; }
  s1s[tid] = s1; s2s[tid] = s2;
  __syncthreads();
  if (tid < 64) {
    float a1 = s1s[tid] + s1s[tid + 64] + s1s[tid + 128] + s1s[tid + 192];
    float a2 = s2s[tid] + s2s[tid + 64] + s2s[tid + 128] + s2s[tid + 192];
    atomicAdd(&stats[tid], a1);
    atomicAdd(&stats[64 + tid], a2);
  }
}
__global__ __launch_bounds__(256) void bn_relu_kernel(
    float* __restrict__ h, const float* __restrict__ stats,
    const float* __restrict__ gamma, const float* __restrict__ beta) {
  size_t i = (size_t)blockIdx.x * 256 + threadIdx.x;
  if (i >= (size_t)TOTAL_ELEMS) return;
  int c = (int)(i & 63);
  float m   = stats[c] * (1.0f / BN_M);
  float var = stats[64 + c] * (1.0f / BN_M) - m * m;
  float inv = rsqrtf(var + 1e-5f);
  float v = (h[i] - m) * inv * gamma[c] + beta[c];
  h[i] = v > 0.0f ? v : 0.0f;
}

// ---------------------------------------------------------------------------
extern "C" void kernel_launch(void* const* d_in, const int* in_sizes, int n_in,
                              void* d_out, int out_size, void* d_ws, size_t ws_size,
                              hipStream_t stream) {
  const float* x     = (const float*)d_in[0];   // (T,N,C) f32
  const int*   ei    = (const int*)d_in[1];     // (2,E)
  const float* c1w   = (const float*)d_in[2];   // (C,C,1,3)
  const float* c1b   = (const float*)d_in[3];
  const float* gw    = (const float*)d_in[4];   // (C,C)
  const float* gb    = (const float*)d_in[5];
  const float* c2w   = (const float*)d_in[6];
  const float* c2b   = (const float*)d_in[7];
  const float* gamma = (const float*)d_in[8];
  const float* beta  = (const float*)d_in[9];
  const int E = in_sizes[1] / 2;

  char* wsp = (char*)d_ws;
  auto carve = [&](size_t bytes) -> char* {
    char* p = wsp; wsp += (bytes + 255) & ~(size_t)255; return p;
  };
  __bf16* h1bf  = (__bf16*)carve((size_t)TOTAL_ELEMS * 2);
  float*  xh    = (float*) carve((size_t)TOTAL_ELEMS * 4);
  float*  agg   = (float*) carve((size_t)TOTAL_ELEMS * 4);
  float*  dinv  = (float*) carve(NN * 4);
  float*  stats = (float*) carve(128 * 4);
  __bf16* wp1   = (__bf16*)carve(3 * 64 * 64 * 2);
  __bf16* wp2   = (__bf16*)carve(3 * 64 * 64 * 2);
  __bf16* gpk   = (__bf16*)carve(64 * 64 * 2);

  // weight packing (tiny)
  pack_kernel<<<(3 * 4096) / 256, 256, 0, stream>>>(c1w, wp1, 3, 1);
  pack_kernel<<<(3 * 4096) / 256, 256, 0, stream>>>(c2w, wp2, 3, 1);
  pack_kernel<<<(1 * 4096) / 256, 256, 0, stream>>>(gw,  gpk, 1, 0);

  // degree / symmetric norm
  deg_init_kernel<<<(NN + 255) / 256, 256, 0, stream>>>(dinv);
  deg_edge_kernel<<<(E + 255) / 256, 256, 0, stream>>>(dinv, ei, E);
  deg_fin_kernel <<<(NN + 255) / 256, 256, 0, stream>>>(dinv);

  // conv1 (f32 in -> bf16 out), GCN linear (bf16 in -> f32 out)
  tconv_kernel<true><<<ROW_TILES / 4, 128, 0, stream>>>(x, wp1, c1b, (void*)h1bf);
  gcn_gemm_kernel   <<<ROW_TILES / 4, 128, 0, stream>>>(h1bf, gpk, xh);

  // scatter-add with gcn bias folded into the init
  agg_init_kernel<<<TOTAL_ELEMS / 256, 256, 0, stream>>>(agg, gb);
  scatter_kernel<<<(((E + NN) * 64) + 255) / 256, 256, 0, stream>>>(agg, xh, ei, dinv, E);

  // conv2 straight into d_out, then BN + ReLU in place
  tconv_kernel<false><<<ROW_TILES / 4, 128, 0, stream>>>(agg, wp2, c2b, d_out);
  stats_init_kernel<<<1, 128, 0, stream>>>(stats);
  stats_kernel<<<512, 256, 0, stream>>>((const float*)d_out, stats);
  bn_relu_kernel<<<TOTAL_ELEMS / 256, 256, 0, stream>>>((float*)d_out, stats, gamma, beta);
}
